// NG_73177652789461
// MI455X (gfx1250) — matrix operand order repaired and checked by reading, not compiled, not run on hardware
//
#include <hip/hip_runtime.h>
#include <hip/hip_bf16.h>

// GEMM: out[32768,256] = x[32768,512] @ W_sel[256,512]^T + b_sel
// W_sel[n,k] = W[1023-n, k],  b_sel[n] = b[1023-n]
// Split-bf16 (hi/lo) WMMA: xh*wh + xl*wh + xh*wl  (~f32 accuracy, f32 accum)
// Double-buffered LDS, one barrier per K-step, hardware v_cvt_pk_bf16_f32 staging.

typedef __attribute__((ext_vector_type(16))) __bf16 v16bf;
typedef __attribute__((ext_vector_type(8)))  float  v8f;

static constexpr int M_DIM = 32768;
static constexpr int K_DIM = 512;
static constexpr int N_DIM = 256;
static constexpr int W_LD  = 1024;   // W row stride
static constexpr int BM = 128;
static constexpr int BN = 64;
static constexpr int BK = 32;
static constexpr int LDA = 40;       // padded LDS row stride (halfwords) -> 80B, conflict-free
static constexpr int KSTEPS = K_DIM / BK;   // 16

// ---- packed f32 -> 2x bf16 (RNE) ----
__device__ __forceinline__ unsigned cvt_pk_bf16(float a, float b) {
#if __has_builtin(__builtin_amdgcn_cvt_pk_bf16_f32)
    auto p = __builtin_amdgcn_cvt_pk_bf16_f32(a, b);
    return __builtin_bit_cast(unsigned, p);
#else
    unsigned d;
    asm("v_cvt_pk_bf16_f32 %0, %1, %2" : "=v"(d) : "v"(a), "v"(b));
    return d;
#endif
}
__device__ __forceinline__ float bf16lo_to_f32(unsigned p) {   // element in [15:0]
    return __builtin_bit_cast(float, p << 16);
}
__device__ __forceinline__ float bf16hi_to_f32(unsigned p) {   // element in [31:16]
    return __builtin_bit_cast(float, p & 0xffff0000u);
}

// convert 4 floats -> hi/lo bf16 pairs, store 8B each to LDS
__device__ __forceinline__ void cvt_store4(float4 v, unsigned short* hp, unsigned short* lp) {
    const unsigned h01 = cvt_pk_bf16(v.x, v.y);
    const unsigned h23 = cvt_pk_bf16(v.z, v.w);
    const float r0 = v.x - bf16lo_to_f32(h01);
    const float r1 = v.y - bf16hi_to_f32(h01);
    const float r2 = v.z - bf16lo_to_f32(h23);
    const float r3 = v.w - bf16hi_to_f32(h23);
    const unsigned l01 = cvt_pk_bf16(r0, r1);
    const unsigned l23 = cvt_pk_bf16(r2, r3);
    uint2 h; h.x = h01; h.y = h23;
    uint2 l; l.x = l01; l.y = l23;
    *reinterpret_cast<uint2*>(hp) = h;   // bf16 order x,y,z,w (little-endian) ✓
    *reinterpret_cast<uint2*>(lp) = l;
}

struct Frag32B { uint4 a, b; };

// 32B fragment as two ds_load_b128, per ISA WMMA VGPR layout
__device__ __forceinline__ v16bf load_frag(const unsigned short* base, int off0, int off1) {
    Frag32B f;
    f.a = *reinterpret_cast<const uint4*>(base + off0);
    f.b = *reinterpret_cast<const uint4*>(base + off1);
    return __builtin_bit_cast(v16bf, f);
}

__device__ __forceinline__ v8f wmma_bf16(v16bf a, v16bf b, v8f c) {
    return __builtin_amdgcn_wmma_f32_16x16x32_bf16(
        false, a, false, b, (short)0, c, false, false);
}

__global__ __launch_bounds__(256)
void NG_gemm_wsel_kernel(const float* __restrict__ x,
                         const float* __restrict__ W,
                         const float* __restrict__ b,
                         float* __restrict__ out) {
    // double-buffered tiles: 61440 B of 320 KB
    __shared__ unsigned short Ah[2][BM * LDA];
    __shared__ unsigned short Al[2][BM * LDA];
    __shared__ unsigned short Bh[2][BN * LDA];
    __shared__ unsigned short Bl[2][BN * LDA];

    const int tid   = threadIdx.x;
    const int lane  = tid & 31;
    const int wave  = tid >> 5;
    const int n0    = blockIdx.x * BN;
    const int m0    = blockIdx.y * BM;
    const int mw    = (wave & 3) * 32;   // 4 waves over M
    const int nw    = (wave >> 2) * 32;  // 2 waves over N
    const int lhalf = lane >> 4;
    const int l16   = lane & 15;

    // ---- staging indices (A: 4 x float4/thread, B: 2 x float4/thread) ----
    const float* xg[4]; int aoff[4];
    #pragma unroll
    for (int u = 0; u < 4; ++u) {
        const int idx = tid + u * 256;         // 0..1023
        const int row = idx >> 3;              // 0..127
        const int kc  = (idx & 7) << 2;        // 0..28
        xg[u]   = x + (size_t)(m0 + row) * K_DIM + kc;
        aoff[u] = row * LDA + kc;
    }
    const float* wgp[2]; int boff[2];
    #pragma unroll
    for (int u = 0; u < 2; ++u) {
        const int idx = tid + u * 256;         // 0..511
        const int n   = idx >> 3;              // 0..63
        const int kc  = (idx & 7) << 2;
        wgp[u]  = W + (size_t)(1023 - (n0 + n)) * W_LD + kc;
        boff[u] = n * LDA + kc;
    }

    // fragment LDS offsets (per ISA layouts), loop-invariant
    int aO[2], bO[2];
    #pragma unroll
    for (int i = 0; i < 2; ++i) aO[i] = (mw + i * 16 + l16) * LDA + lhalf * 8;
    #pragma unroll
    for (int j = 0; j < 2; ++j) bO[j] = (nw + j * 16 + l16) * LDA + lhalf * 8;

    v8f acc[2][2];
    #pragma unroll
    for (int i = 0; i < 2; ++i)
        #pragma unroll
        for (int j = 0; j < 2; ++j)
            acc[i][j] = v8f{};

    // ---- prologue: stage K-step 0 into buffer 0 ----
    {
        float4 ga[4], gb[2];
        #pragma unroll
        for (int u = 0; u < 4; ++u) ga[u] = *reinterpret_cast<const float4*>(xg[u]);
        #pragma unroll
        for (int u = 0; u < 2; ++u) gb[u] = *reinterpret_cast<const float4*>(wgp[u]);
        #pragma unroll
        for (int u = 0; u < 4; ++u) cvt_store4(ga[u], &Ah[0][aoff[u]], &Al[0][aoff[u]]);
        #pragma unroll
        for (int u = 0; u < 2; ++u) cvt_store4(gb[u], &Bh[0][boff[u]], &Bl[0][boff[u]]);
        __syncthreads();
    }

    for (int kt = 0; kt < KSTEPS; ++kt) {
        const int  cur  = kt & 1;
        const int  nxt  = cur ^ 1;
        const bool more = (kt + 1 < KSTEPS);

        // 1) issue next tile's global loads first: whole WMMA phase to hide latency
        float4 ga[4], gb[2];
        if (more) {
            const int k0n = (kt + 1) * BK;
            #pragma unroll
            for (int u = 0; u < 4; ++u) ga[u] = *reinterpret_cast<const float4*>(xg[u] + k0n);
            #pragma unroll
            for (int u = 0; u < 2; ++u) gb[u] = *reinterpret_cast<const float4*>(wgp[u] + k0n);
        }

        // 2) fragments from current buffer
        v16bf ah[2], al2[2], bh[2], bl2[2];
        #pragma unroll
        for (int i = 0; i < 2; ++i) {
            ah[i]  = load_frag(Ah[cur], aO[i], aO[i] + 16);
            al2[i] = load_frag(Al[cur], aO[i], aO[i] + 16);
        }
        #pragma unroll
        for (int j = 0; j < 2; ++j) {
            bh[j]  = load_frag(Bh[cur], bO[j], bO[j] + 16);
            bl2[j] = load_frag(Bl[cur], bO[j], bO[j] + 16);
        }

        // 3) 12 WMMAs, term-major: 3 independent ops between updates of each acc
        #pragma unroll
        for (int i = 0; i < 2; ++i)
            #pragma unroll
            for (int j = 0; j < 2; ++j)
                acc[i][j] = wmma_bf16(ah[i], bh[j], acc[i][j]);
        #pragma unroll
        for (int i = 0; i < 2; ++i)
            #pragma unroll
            for (int j = 0; j < 2; ++j)
                acc[i][j] = wmma_bf16(al2[i], bh[j], acc[i][j]);
        #pragma unroll
        for (int i = 0; i < 2; ++i)
            #pragma unroll
            for (int j = 0; j < 2; ++j)
                acc[i][j] = wmma_bf16(ah[i], bl2[j], acc[i][j]);

        // 4) stage next tile into the other buffer, single barrier per K-step
        if (more) {
            #pragma unroll
            for (int u = 0; u < 4; ++u) cvt_store4(ga[u], &Ah[nxt][aoff[u]], &Al[nxt][aoff[u]]);
            #pragma unroll
            for (int u = 0; u < 2; ++u) cvt_store4(gb[u], &Bh[nxt][boff[u]], &Bl[nxt][boff[u]]);
            __syncthreads();
        }
    }

    // ---- epilogue: D layout -> global, add bias b[1023 - n] ----
    float bias[2];
    #pragma unroll
    for (int j = 0; j < 2; ++j)
        bias[j] = b[1023 - (n0 + nw + j * 16 + l16)];

    #pragma unroll
    for (int i = 0; i < 2; ++i) {
        #pragma unroll
        for (int j = 0; j < 2; ++j) {
            const int nG = n0 + nw + j * 16 + l16;
            const int mG = m0 + mw + i * 16 + lhalf * 8;
            float* op = out + (size_t)mG * N_DIM + nG;
            #pragma unroll
            for (int r = 0; r < 8; ++r)
                op[(size_t)r * N_DIM] = acc[i][j][r] + bias[j];
        }
    }
}

extern "C" void kernel_launch(void* const* d_in, const int* in_sizes, int n_in,
                              void* d_out, int out_size, void* d_ws, size_t ws_size,
                              hipStream_t stream) {
    const float* x = (const float*)d_in[0];   // [32768, 512]
    const float* W = (const float*)d_in[1];   // [1024, 1024]
    const float* b = (const float*)d_in[2];   // [1024]
    float* out = (float*)d_out;               // [32768, 256]
    (void)in_sizes; (void)n_in; (void)out_size; (void)d_ws; (void)ws_size;

    dim3 grid(N_DIM / BN, M_DIM / BM);        // (4, 256)
    dim3 block(256);
    NG_gemm_wsel_kernel<<<grid, block, 0, stream>>>(x, W, b, out);
}
